// BasicGNNClassifier_6571299963161
// MI455X (gfx1250) — matrix operand order repaired and checked by reading, not compiled, not run on hardware
//
#include <hip/hip_runtime.h>
#include <hip/hip_bf16.h>
#include <stdint.h>

#define N_NODES   50000
#define N_EDGES   800000
#define DFEAT     128
#define N_CLASSES 16
#define N_GRAPHS  256

typedef __attribute__((ext_vector_type(16))) __bf16 v16bf;
typedef __attribute__((ext_vector_type(8)))  __bf16 v8bf;
typedef __attribute__((ext_vector_type(8)))  float  v8f;

union bfrag { v16bf v; v8bf h[2]; };

__device__ __forceinline__ __bf16 f2bf(float f) {
    union { float f; uint32_t u; } v; v.f = f;
    uint32_t r = v.u + 0x7FFFu + ((v.u >> 16) & 1u);   // round-to-nearest-even
    uint16_t h = (uint16_t)(r >> 16);
    return __builtin_bit_cast(__bf16, h);
}

// ---------------------------------------------------------------------------
// Hout[M x 128] = A[M x 128](bf16) @ B, where BT[n][k] is B pre-transposed
// bf16.  One wave = one 16x16 f32 tile via v_wmma_f32_16x16x32_bf16.
// Fragment halves are contiguous 8-element runs -> pure b128 loads, no VALU.
// ---------------------------------------------------------------------------
__global__ void gemm_bf16_wmma(const __bf16* __restrict__ A,
                               const __bf16* __restrict__ BT,
                               float* __restrict__ Hout, int M) {
    const int wave = threadIdx.x >> 5;
    const int lane = threadIdx.x & 31;
    const int tile = blockIdx.x * (blockDim.x >> 5) + wave;
    const int tilesN = DFEAT / 16;                 // 8
    const int mi = tile / tilesN, ni = tile % tilesN;
    const int m0 = mi * 16, n0 = ni * 16;
    if (m0 >= M) return;

    const int row = lane & 15;                     // M for A, N for B/C/D
    const int hi  = lane >> 4;                     // half-wave selector

    const __bf16* arow = A  + (size_t)(m0 + row) * DFEAT;
    const __bf16* brow = BT + (size_t)(n0 + row) * DFEAT;

    v8f acc = {};
#pragma unroll
    for (int kc = 0; kc < DFEAT; kc += 32) {
        bfrag a, b;
        // A 16x32: lane holds K = kc+8*hi+[0..8) and kc+16+8*hi+[0..8)
        a.h[0] = *(const v8bf*)(arow + kc + 8 * hi);
        a.h[1] = *(const v8bf*)(arow + kc + 16 + 8 * hi);
        // B 32x16: lane (col n0+row) holds K = kc+8*hi+[0..8) and kc+16+8*hi+[0..8)
        b.h[0] = *(const v8bf*)(brow + kc + 8 * hi);
        b.h[1] = *(const v8bf*)(brow + kc + 16 + 8 * hi);
        acc = __builtin_amdgcn_wmma_f32_16x16x32_bf16(
                  false, a.v, false, b.v, (short)0, acc, false, false);
    }
#pragma unroll
    for (int v = 0; v < 8; ++v)
        Hout[(size_t)(m0 + v + hi * 8) * DFEAT + (n0 + row)] = acc[v];
}

// ---------------------------------------------------------------------------
// Conversions
// ---------------------------------------------------------------------------
__global__ void cvt_f32_bf16(const float* __restrict__ in,
                             __bf16* __restrict__ out, int n) {
    int i = blockIdx.x * blockDim.x + threadIdx.x;
    if (i < n) out[i] = f2bf(in[i]);
}
__global__ void transpose_w_bf16(const float* __restrict__ W,
                                 __bf16* __restrict__ WT) {
    int t = blockIdx.x * blockDim.x + threadIdx.x;
    if (t < DFEAT * DFEAT) {
        int k = t / DFEAT, n = t % DFEAT;
        WT[(size_t)n * DFEAT + k] = f2bf(W[t]);
    }
}

// ---------------------------------------------------------------------------
// Degree / normalization
// ---------------------------------------------------------------------------
__global__ void init_deg(float* deg) {
    int i = blockIdx.x * blockDim.x + threadIdx.x;
    if (i < N_NODES) deg[i] = 1.0f;                // self-loop contribution
}
__global__ void deg_kernel(const int* __restrict__ dst, float* deg) {
    int e = blockIdx.x * blockDim.x + threadIdx.x;
    if (e < N_EDGES) atomicAdd(&deg[dst[e]], 1.0f);
}
__global__ void dinv_kernel(const float* __restrict__ deg, float* dinv) {
    int i = blockIdx.x * blockDim.x + threadIdx.x;
    if (i < N_NODES) {
        float d = deg[i];
        dinv[i] = (d > 0.0f) ? rsqrtf(d) : 0.0f;
    }
}

// ---------------------------------------------------------------------------
// Edge scatter: one wave per edge, each lane handles 4 channels (float4).
// agg[dst] += H[src] * (dinv[src]*dinv[dst])
// ---------------------------------------------------------------------------
__global__ void scatter_kernel(const float* __restrict__ H,
                               const int* __restrict__ srcI,
                               const int* __restrict__ dstI,
                               const float* __restrict__ dinv,
                               float* __restrict__ agg) {
    const long tid = (long)blockIdx.x * blockDim.x + threadIdx.x;
    const int e = (int)(tid >> 5);
    if (e >= N_EDGES) return;
    const int lane = (int)(tid & 31);
    const int s = srcI[e], d = dstI[e];
    const float n = dinv[s] * dinv[d];
    const float4 v = ((const float4*)(H + (size_t)s * DFEAT))[lane];
    float* ar = agg + (size_t)d * DFEAT + lane * 4;
    atomicAdd(ar + 0, v.x * n);
    atomicAdd(ar + 1, v.y * n);
    atomicAdd(ar + 2, v.z * n);
    atomicAdd(ar + 3, v.w * n);
}

// ---------------------------------------------------------------------------
// out = agg + H*dinv^2 (self-loop) + bias  [+ReLU]; optional bf16 copy for
// the next layer's WMMA input.  In-place safe (elementwise).
// ---------------------------------------------------------------------------
__global__ void fuse_kernel(const float* __restrict__ agg,
                            const float* __restrict__ H,
                            const float* __restrict__ dinv,
                            const float* __restrict__ bias,
                            float* __restrict__ out,
                            __bf16* __restrict__ out_bf, int do_relu) {
    const int idx = blockIdx.x * blockDim.x + threadIdx.x;
    if (idx >= N_NODES * DFEAT) return;
    const int i = idx / DFEAT, c = idx % DFEAT;
    const float di = dinv[i];
    float v = agg[idx] + H[idx] * di * di + bias[c];
    if (do_relu) v = fmaxf(v, 0.0f);
    out[idx] = v;
    if (out_bf) out_bf[idx] = f2bf(v);
}

// ---------------------------------------------------------------------------
// Mean-pool accumulation: one wave per node (float4 per lane).
// ---------------------------------------------------------------------------
__global__ void pool_kernel(const float* __restrict__ H,
                            const int* __restrict__ batch,
                            float* __restrict__ sums, float* __restrict__ cnt) {
    const long tid = (long)blockIdx.x * blockDim.x + threadIdx.x;
    const int i = (int)(tid >> 5);
    if (i >= N_NODES) return;
    const int lane = (int)(tid & 31);
    const int g = batch[i];
    const float4 v = ((const float4*)(H + (size_t)i * DFEAT))[lane];
    float* s = sums + (size_t)g * DFEAT + lane * 4;
    atomicAdd(s + 0, v.x);
    atomicAdd(s + 1, v.y);
    atomicAdd(s + 2, v.z);
    atomicAdd(s + 3, v.w);
    if (lane == 0) atomicAdd(cnt + g, 1.0f);
}

// ---------------------------------------------------------------------------
// Classifier: out[g,c] = (sums[g]/max(cnt,1)) . Wc[:,c] + bc[c]
// ---------------------------------------------------------------------------
__global__ void classifier_kernel(const float* __restrict__ sums,
                                  const float* __restrict__ cnt,
                                  const float* __restrict__ Wc,
                                  const float* __restrict__ bc,
                                  float* __restrict__ out) {
    const int t = blockIdx.x * blockDim.x + threadIdx.x;
    if (t >= N_GRAPHS * N_CLASSES) return;
    const int g = t / N_CLASSES, c = t % N_CLASSES;
    const float inv = 1.0f / fmaxf(cnt[g], 1.0f);
    float acc = bc[c];
#pragma unroll 4
    for (int k = 0; k < DFEAT; ++k)
        acc += sums[(size_t)g * DFEAT + k] * inv * Wc[k * N_CLASSES + c];
    out[t] = acc;
}

extern "C" void kernel_launch(void* const* d_in, const int* in_sizes, int n_in,
                              void* d_out, int out_size, void* d_ws, size_t ws_size,
                              hipStream_t stream) {
    (void)in_sizes; (void)n_in; (void)out_size; (void)ws_size;
    const float* x   = (const float*)d_in[0];
    const int*   ei  = (const int*)d_in[1];
    const int*   bat = (const int*)d_in[2];
    const float* W1  = (const float*)d_in[3];
    const float* b1  = (const float*)d_in[4];
    const float* W2  = (const float*)d_in[5];
    const float* b2  = (const float*)d_in[6];
    const float* Wc  = (const float*)d_in[7];
    const float* bc  = (const float*)d_in[8];
    float* out = (float*)d_out;

    char* ws = (char*)d_ws;
    const size_t featBytes   = (size_t)N_NODES * DFEAT * sizeof(float);   // 25.6 MB
    const size_t featBytesBf = (size_t)N_NODES * DFEAT * sizeof(__bf16);  // 12.8 MB
    float*  bufA = (float*)ws;
    float*  bufB = (float*)(ws + featBytes);
    __bf16* Xb   = (__bf16*)(ws + 2 * featBytes);          // bf16 input / activations
    float*  deg  = (float*)(ws + 2 * featBytes + featBytesBf);
    float*  dinv = deg + N_NODES;
    float*  sums = dinv + N_NODES;
    float*  cnt  = sums + (size_t)N_GRAPHS * DFEAT;
    __bf16* WT1  = (__bf16*)(cnt + N_GRAPHS);
    __bf16* WT2  = WT1 + DFEAT * DFEAT;

    const int* srcI = ei;            // edge_index[0]
    const int* dstI = ei + N_EDGES;  // edge_index[1]

    // --- normalization ---
    init_deg<<<(N_NODES + 255) / 256, 256, 0, stream>>>(deg);
    deg_kernel<<<(N_EDGES + 255) / 256, 256, 0, stream>>>(dstI, deg);
    dinv_kernel<<<(N_NODES + 255) / 256, 256, 0, stream>>>(deg, dinv);

    // --- precision prep (one-shot conversions) ---
    cvt_f32_bf16<<<(N_NODES * DFEAT + 255) / 256, 256, 0, stream>>>(x, Xb, N_NODES * DFEAT);
    transpose_w_bf16<<<(DFEAT * DFEAT + 255) / 256, 256, 0, stream>>>(W1, WT1);
    transpose_w_bf16<<<(DFEAT * DFEAT + 255) / 256, 256, 0, stream>>>(W2, WT2);

    const int tileBlocks = (N_NODES / 16) * (DFEAT / 16) / 8;  // 3125 blocks, 8 waves each
    const int edgeBlocks = (int)(((long)N_EDGES * 32 + 255) / 256);
    const int featBlocks = (N_NODES * DFEAT + 255) / 256;

    // --- layer 1 ---
    gemm_bf16_wmma<<<tileBlocks, 256, 0, stream>>>(Xb, WT1, bufA, N_NODES);
    hipMemsetAsync(bufB, 0, featBytes, stream);
    scatter_kernel<<<edgeBlocks, 256, 0, stream>>>(bufA, srcI, dstI, dinv, bufB);
    // H1act -> bufA (f32) and Xb (bf16, reused as GEMM2 input)
    fuse_kernel<<<featBlocks, 256, 0, stream>>>(bufB, bufA, dinv, b1, bufA, Xb, 1);

    // --- layer 2 ---
    gemm_bf16_wmma<<<tileBlocks, 256, 0, stream>>>(Xb, WT2, bufB, N_NODES);
    hipMemsetAsync(bufA, 0, featBytes, stream);
    scatter_kernel<<<edgeBlocks, 256, 0, stream>>>(bufB, srcI, dstI, dinv, bufA);
    fuse_kernel<<<featBlocks, 256, 0, stream>>>(bufA, bufB, dinv, b2, bufA, (__bf16*)nullptr, 0);

    // --- pool + classify ---
    hipMemsetAsync(sums, 0, ((size_t)N_GRAPHS * DFEAT + N_GRAPHS) * sizeof(float), stream);
    pool_kernel<<<(int)(((long)N_NODES * 32 + 255) / 256), 256, 0, stream>>>(bufA, bat, sums, cnt);
    classifier_kernel<<<(N_GRAPHS * N_CLASSES + 255) / 256, 256, 0, stream>>>(sums, cnt, Wc, bc, out);
}
